// MasteringModel_Phase1B_76141180223829
// MI455X (gfx1250) — matrix operand order repaired and checked by reading, not compiled, not run on hardware
//
#include <hip/hip_runtime.h>
#include <math.h>

// ---------------------------------------------------------------------------
// MI455X (gfx1250) mastering model.
// Convs -> bf16 WMMA 16x16x32 implicit GEMM, f32 accum, fused epilogues.
//   * weights pre-packed to WMMA A-fragment order (bf16, padded-K) -> b128 loads
//   * padded-K indexing: ci/kq compile-time per element, no div/mod, no ragged
//   * bf16 activations everywhere; epilogue writes f32 and/or bf16 twins
//   * edge tiles: clamped address + u16 AND-mask (unconditional batched loads)
// Biquad EQ -> single-pass 5-stage cascade (16 serial streams).
// ---------------------------------------------------------------------------

typedef __attribute__((ext_vector_type(16))) __bf16 v16bf;
typedef __attribute__((ext_vector_type(8)))  float  v8f;

#define BN_EPS 1e-5f

__device__ __host__ inline int a_koff(int h, int e) {   // contiguous k = 16h+e order used here for A and B
  return 16 * h + e;
}
__device__ inline int imin(int a, int b) { return a < b ? a : b; }
__device__ inline int imax(int a, int b) { return a > b ? a : b; }

// --------------------- weight pre-pack (f32 -> bf16 frags) -----------------
// K-dim re-indexed as kk = ci*KP + kq (KP power of two >= K); zero padding for
// kq >= K is baked in. Layout: wp[(((mt*Ksteps + ks)*32 + lane)*16 + e].
__global__ void pack_weights_kernel(const float* __restrict__ w, __bf16* __restrict__ wp,
                                    long total, int Ksteps, int Cin, int K, int KP)
{
  const long idx = (long)blockIdx.x * blockDim.x + threadIdx.x;
  if (idx >= total) return;
  const int e    = (int)(idx & 15);
  const int lane = (int)((idx >> 4) & 31);
  long r = idx >> 9;
  const int ks = (int)(r % Ksteps);
  const int mt = (int)(r / Ksteps);
  const int h = lane >> 4, ln = lane & 15;
  const int kk = ks * 32 + a_koff(h, e);
  const int ci = kk / KP;
  const int kq = kk - ci * KP;
  const int m  = mt * 16 + ln;
  float v = 0.f;
  if (ci < Cin && kq < K) v = w[((long)m * Cin + ci) * K + kq];
  wp[idx] = (__bf16)v;
}

// ------------------------- WMMA implicit-GEMM conv -------------------------
// One wave -> 16(cout) x 16(time) tile; 8 waves/block share one cout tile.
// Requires: Cout % 16 == 0, Tout % 128 == 0, (Cin*KP) % 32 == 0 (all true).
template <int K, int KP>
__global__ __launch_bounds__(256) void conv1d_wmma_kernel(
    const __bf16* __restrict__ x, const __bf16* __restrict__ wp,
    const float* __restrict__ bias,
    const float* __restrict__ bng, const float* __restrict__ bnb,
    const float* __restrict__ bnm, const float* __restrict__ bnv,
    const float* __restrict__ fg,  const float* __restrict__ fb,
    const float* __restrict__ resid,
    float* __restrict__ yf, __bf16* __restrict__ yb,
    int B, int Cin, int Tin, int Cout, int Tout,
    int stride, int dil, int pad, int relu_mode)
{
  const int lane = threadIdx.x & 31;
  const int wave = threadIdx.x >> 5;
  const int mTiles   = Cout >> 4;
  const int ntGroups = Tout >> 7;

  unsigned bid = blockIdx.x;
  const int ng = (int)(bid % ntGroups); bid /= ntGroups;
  const int mt = (int)(bid % mTiles);
  const int b  = (int)(bid / mTiles);

  const int m0 = mt << 4;
  const int n0 = ((ng << 3) + wave) << 4;
  const int Ksteps = (Cin * KP) >> 5;
  const int ciStep = 32 / KP;          // ci's consumed per K-step
  const int ciHalf = 16 / KP;          // ci offset of upper half-wave

  const int h  = lane >> 4;
  const int ln = lane & 15;
  const int tb = (n0 + ln) * stride - pad;               // per-lane time base
  const __bf16* xp = x + (long)b * Cin * Tin;            // batch base

  const int tLo = n0 * stride - pad;
  const int tHi = (n0 + 15) * stride + (K - 1) * dil - pad;
  const bool fast = (tLo >= 0) && (tHi < Tin);

  v8f acc = {0.f, 0.f, 0.f, 0.f, 0.f, 0.f, 0.f, 0.f};
  const __bf16* ap = wp + (((long)mt * Ksteps) * 32 + lane) * 16;

  if (fast) {
    for (int ks = 0; ks < Ksteps; ++ks, ap += 512) {
      const v16bf afrag = *reinterpret_cast<const v16bf*>(ap);
      v16bf bfrag;
      const int cibase = ks * ciStep + h * ciHalf;
#pragma unroll
      for (int e = 0; e < 16; ++e) {
        const int kq = e & (KP - 1);                     // compile-time
        if (kq < K) {                                    // compile-time
          const int ci = cibase + (e / KP);              // compile-time shift
          bfrag[e] = xp[(long)ci * Tin + (tb + kq * dil)];
        } else {
          bfrag[e] = (__bf16)0.f;
        }
      }
      acc = __builtin_amdgcn_wmma_f32_16x16x32_bf16(false, afrag, false, bfrag,
                                                    (short)0, acc, false, false);
    }
  } else {
    // Edge tiles: clamped in-bounds address + AND-mask of the raw u16 value.
    // The AND uses the load result unconditionally, so loads stay batched.
    const unsigned short* xu = reinterpret_cast<const unsigned short*>(xp);
    for (int ks = 0; ks < Ksteps; ++ks, ap += 512) {
      const v16bf afrag = *reinterpret_cast<const v16bf*>(ap);
      v16bf bfrag;
      const int cibase = ks * ciStep + h * ciHalf;
#pragma unroll
      for (int e = 0; e < 16; ++e) {
        const int kq = e & (KP - 1);                     // compile-time
        if (kq < K) {                                    // compile-time
          const int ci = cibase + (e / KP);
          const int t  = tb + kq * dil;                  // loop-invariant per e
          const int tc = imin(imax(t, 0), Tin - 1);
          const unsigned short msk =
              (unsigned short)(((t >= 0) && (t < Tin)) ? 0xFFFFu : 0u);
          const unsigned short raw =
              (unsigned short)(xu[(long)ci * Tin + tc] & msk);
          bfrag[e] = __builtin_bit_cast(__bf16, raw);
        } else {
          bfrag[e] = (__bf16)0.f;
        }
      }
      acc = __builtin_amdgcn_wmma_f32_16x16x32_bf16(false, afrag, false, bfrag,
                                                    (short)0, acc, false, false);
    }
  }

  // Epilogue; C/D layout: VGPR i -> M = i + 8h, N = lane&15
  const int n = n0 + ln;
#pragma unroll
  for (int i = 0; i < 8; ++i) {
    const int m = m0 + i + 8 * h;
    float v = acc[i] + bias[m];
    if (bng) {
      const float sc = bng[m] * rsqrtf(bnv[m] + BN_EPS);
      v = v * sc + (bnb[m] - bnm[m] * sc);
    }
    if (relu_mode == 1) v = fmaxf(v, 0.f);
    if (fg) v = fg[b * Cout + m] * v + fb[b * Cout + m];
    const long oidx = ((long)b * Cout + m) * Tout + n;
    if (resid) v += resid[oidx];
    if (relu_mode == 2) v = fmaxf(v, 0.f);
    if (yf) yf[oidx] = v;
    if (yb) yb[oidx] = (__bf16)v;
  }
}

// ----------------------------- small kernels -------------------------------
__global__ void cast_bf16_kernel(const float* __restrict__ x, __bf16* __restrict__ y, long n)
{
  const long idx = (long)blockIdx.x * blockDim.x + threadIdx.x;
  if (idx < n) y[idx] = (__bf16)x[idx];
}

__global__ void linear_kernel(const float* __restrict__ x, const float* __restrict__ w,
                              const float* __restrict__ b, float* __restrict__ y,
                              int B, int Cin, int Cout, int act, float scale, float offs)
{
  const int i = blockIdx.x * blockDim.x + threadIdx.x;
  if (i >= B * Cout) return;
  const int bb = i / Cout, o = i - bb * Cout;
  const float* xr = x + (long)bb * Cin;
  const float* wr = w + (long)o * Cin;
  float s = b[o];
  for (int c = 0; c < Cin; ++c) s += xr[c] * wr[c];
  if (act == 1)      s = fmaxf(s, 0.f);
  else if (act == 2) s = (1.f / (1.f + expf(-s))) * scale + offs;
  else if (act == 3) s = tanhf(s) * scale + offs;
  y[i] = s;
}

__global__ void mean_kernel(const float* __restrict__ x, float* __restrict__ y,
                            int BC, int T)
{
  const int i = blockIdx.x * blockDim.x + threadIdx.x;
  if (i >= BC) return;
  const float* p = x + (long)i * T;
  float s = 0.f;
  for (int t = 0; t < T; ++t) s += p[t];
  y[i] = s / (float)T;
}

// f32 in -> bf16 out (pooled tensors only feed convs)
__global__ void avgpool2_kernel(const float* __restrict__ x, __bf16* __restrict__ y,
                                long total, int Lo)
{
  const long idx = (long)blockIdx.x * blockDim.x + threadIdx.x;
  if (idx >= total) return;
  const int  t  = (int)(idx % Lo);
  const long bc = idx / Lo;
  const float* p = x + bc * (long)(2 * Lo);
  y[idx] = (__bf16)(0.5f * (p[2 * t] + p[2 * t + 1]));
}

// f32 in -> bf16 out into concat slot (concat buffers only feed convs)
__global__ void interp_concat_kernel(const float* __restrict__ x, __bf16* __restrict__ out,
                                     long total, int C, int Tin, int Tout,
                                     int Ctot, int c0)
{
  const long idx = (long)blockIdx.x * blockDim.x + threadIdx.x;
  if (idx >= total) return;
  const int  t = (int)(idx % Tout);
  long r = idx / Tout;
  const int c = (int)(r % C);
  const int b = (int)(r / C);
  float pos = ((float)t + 0.5f) * ((float)Tin / (float)Tout) - 0.5f;
  pos = fminf(fmaxf(pos, 0.f), (float)(Tin - 1));
  const int i0 = (int)floorf(pos);
  const int i1 = min(i0 + 1, Tin - 1);
  const float wgt = pos - (float)i0;
  const float* xr = x + ((long)b * C + c) * Tin;
  out[((long)b * Ctot + (c0 + c)) * Tout + t] = (__bf16)(xr[i0] * (1.f - wgt) + xr[i1] * wgt);
}

__global__ void pconv_kernel(const float* __restrict__ x, const float* __restrict__ w,
                             const float* __restrict__ bias, float* __restrict__ y,
                             long total, int Cin, int Cout, int T)
{
  const long idx = (long)blockIdx.x * blockDim.x + threadIdx.x;
  if (idx >= total) return;
  const int  t = (int)(idx % T);
  long r = idx / T;
  const int o = (int)(r % Cout);
  const int b = (int)(r / Cout);
  float s = bias[o];
  const float* xbp = x + (long)b * Cin * T + t;
  const float* wr  = w + (long)o * Cin;
  for (int ci = 0; ci < Cin; ++ci) s += wr[ci] * xbp[(long)ci * T];
  y[idx] = s;
}

__global__ void eq_coef_kernel(const float* __restrict__ freqs, const float* __restrict__ gains,
                               const float* __restrict__ qs, float* __restrict__ coef, int n)
{
  const int i = blockIdx.x * blockDim.x + threadIdx.x;
  if (i >= n) return;
  const float w0 = 2.f * 3.14159265358979323846f * freqs[i] / 44100.f;
  const float alpha = sinf(w0) / (2.f * qs[i]);
  const float A  = powf(10.f, gains[i] * 0.025f);
  const float b0 = 1.f + alpha * A;
  const float b1 = -2.f * cosf(w0);
  const float b2 = 1.f - alpha * A;
  const float a0 = 1.f + alpha / A;
  const float a2 = 1.f - alpha / A;
  const float inv = 1.f / a0;
  coef[i * 5 + 0] = b0 * inv;
  coef[i * 5 + 1] = b1 * inv;
  coef[i * 5 + 2] = b2 * inv;
  coef[i * 5 + 3] = b1 * inv;   // a1 == b1 for peaking EQ
  coef[i * 5 + 4] = a2 * inv;
}

__global__ void biquad_cascade_kernel(const float* __restrict__ audio,
                                      const float* __restrict__ coef,
                                      float* __restrict__ eq, int B, int T)
{
  const int i = threadIdx.x;
  if (i >= B * 2) return;
  const int b = i >> 1;
  float cb0[5], cb1[5], cb2[5], ca1[5], ca2[5];
#pragma unroll
  for (int s = 0; s < 5; ++s) {
    const float* c = coef + ((long)b * 5 + s) * 5;
    cb0[s] = c[0]; cb1[s] = c[1]; cb2[s] = c[2]; ca1[s] = c[3]; ca2[s] = c[4];
  }
  float x1[5] = {0, 0, 0, 0, 0}, x2[5] = {0, 0, 0, 0, 0};
  float y1[5] = {0, 0, 0, 0, 0}, y2[5] = {0, 0, 0, 0, 0};
  const float* in  = audio + (long)i * T;
  float*       out = eq    + (long)i * T;
  for (int t = 0; t < T; ++t) {
    float s = in[t];
#pragma unroll
    for (int st = 0; st < 5; ++st) {
      const float yv = cb0[st] * s + cb1[st] * x1[st] + cb2[st] * x2[st]
                     - ca1[st] * y1[st] - ca2[st] * y2[st];
      x2[st] = x1[st]; x1[st] = s;
      y2[st] = y1[st]; y1[st] = yv;
      s = yv;
    }
    out[t] = s;
  }
}

__global__ void add_kernel(const float* __restrict__ a, const float* __restrict__ b,
                           float* __restrict__ y, long n)
{
  const long idx = (long)blockIdx.x * blockDim.x + threadIdx.x;
  if (idx < n) y[idx] = a[idx] + b[idx];
}

// ------------------------------ host side ----------------------------------
static inline unsigned gridFor(long n) { return (unsigned)((n + 255) / 256); }

extern "C" void kernel_launch(void* const* d_in, const int* in_sizes, int n_in,
                              void* d_out, int out_size, void* d_ws, size_t ws_size,
                              hipStream_t stream)
{
  (void)in_sizes; (void)out_size; (void)ws_size;
  const float* IN[160];
  for (int i = 0; i < n_in && i < 160; ++i) IN[i] = (const float*)d_in[i];
  const float* audio = IN[0];

  const int B = 8, T = 65536;
  float* outp  = (float*)d_out;
  float* z     = outp + (long)B * 2 * T;
  float* freqs = z + B * 512;
  float* gains = freqs + B * 5;
  float* qsb   = gains + B * 5;

  float* ws = (float*)d_ws;
  size_t off = 0;
  auto alloc  = [&](size_t n) { float* p = ws + off; off += (n + 63) & ~(size_t)63; return p; };
  auto allocB = [&](size_t n) { return (__bf16*)alloc((n + 1) / 2); };

  // ---- persistent scratch ----
  float*  eq     = alloc((size_t)B * 2 * T);
  float*  coefs  = alloc(B * 5 * 5);
  float*  zpre   = alloc(B * 512);
  float*  ph1    = alloc(B * 256);
  float*  ph2    = alloc(B * 256);
  __bf16* audioB = allocB((size_t)B * 2 * T);
  const int wubC[7]    = {32, 64, 128, 256, 128, 64, 32};
  const int wubBase[7] = {77, 87, 97, 107, 117, 127, 137};
  float* filmG[7]; float* filmB[7];
  for (int i = 0; i < 7; ++i) { filmG[i] = alloc(B * wubC[i]); filmB[i] = alloc(B * wubC[i]); }

  cast_bf16_kernel<<<gridFor((long)B * 2 * T), 256, 0, stream>>>(audio, audioB, (long)B * 2 * T);

  // conv wrapper: pre-pack weights, then run the WMMA conv (templated on K/KP)
  auto conv1d = [&](const __bf16* x, const float* w, const float* bias,
                    const float* bng, const float* bnb, const float* bnm, const float* bnv,
                    const float* fg, const float* fb, const float* resid,
                    float* yf, __bf16* yb,
                    int Bn, int Cin, int Tin, int Cout, int Tout,
                    int K, int stride, int dil, int pad, int relu_mode) {
    const int KP = (K == 15) ? 16 : (K == 3) ? 4 : 1;
    const int Ksteps = (Cin * KP) / 32;
    const int mTiles = Cout / 16;
    const long packElems = (long)mTiles * Ksteps * 512;
    __bf16* wp = allocB((size_t)packElems);
    pack_weights_kernel<<<gridFor(packElems), 256, 0, stream>>>(w, wp, packElems, Ksteps, Cin, K, KP);
    const long blocks = (long)Bn * mTiles * (Tout / 128);
    if (K == 15)
      conv1d_wmma_kernel<15, 16><<<dim3((unsigned)blocks), dim3(256), 0, stream>>>(
          x, wp, bias, bng, bnb, bnm, bnv, fg, fb, resid, yf, yb,
          Bn, Cin, Tin, Cout, Tout, stride, dil, pad, relu_mode);
    else if (K == 3)
      conv1d_wmma_kernel<3, 4><<<dim3((unsigned)blocks), dim3(256), 0, stream>>>(
          x, wp, bias, bng, bnb, bnm, bnv, fg, fb, resid, yf, yb,
          Bn, Cin, Tin, Cout, Tout, stride, dil, pad, relu_mode);
    else
      conv1d_wmma_kernel<1, 1><<<dim3((unsigned)blocks), dim3(256), 0, stream>>>(
          x, wp, bias, bng, bnb, bnm, bnv, fg, fb, resid, yf, yb,
          Bn, Cin, Tin, Cout, Tout, stride, dil, pad, relu_mode);
  };

  const size_t base0 = off;

  // ================= encoder =================
  __bf16* s1b = allocB((size_t)B * 64 * 16384);
  conv1d(audioB, IN[1], IN[2], IN[3], IN[4], IN[5], IN[6],
         nullptr, nullptr, nullptr, nullptr, s1b,
         B, 2, 65536, 64, 16384, 15, 4, 1, 7, 1);
  float*  s2f = alloc((size_t)B * 128 * 4096);
  __bf16* s2b = allocB((size_t)B * 128 * 4096);
  conv1d(s1b, IN[7], IN[8], IN[9], IN[10], IN[11], IN[12],
         nullptr, nullptr, nullptr, s2f, s2b,
         B, 64, 16384, 128, 4096, 15, 4, 1, 7, 1);

  struct TB { int base, cin, cout, d, resW, resB; bool hasRes; };
  const TB tb[4] = {{13, 128, 128, 1, 0, 0, false},
                    {25, 128, 256, 2, 37, 38, true},
                    {39, 256, 256, 4, 0, 0, false},
                    {51, 256, 512, 8, 63, 64, true}};
  const float*  curF = s2f;
  const __bf16* curB = s2b;
  for (int k = 0; k < 4; ++k) {
    const TB& t = tb[k];
    const float* res = curF;
    if (t.hasRes) {
      float* rbf = alloc((size_t)B * t.cout * 4096);
      conv1d(curB, IN[t.resW], IN[t.resB], nullptr, nullptr, nullptr, nullptr,
             nullptr, nullptr, nullptr, rbf, nullptr,
             B, t.cin, 4096, t.cout, 4096, 1, 1, 1, 0, 0);
      res = rbf;
    }
    __bf16* hB = allocB((size_t)B * t.cout * 4096);
    conv1d(curB, IN[t.base], IN[t.base + 1],
           IN[t.base + 2], IN[t.base + 3], IN[t.base + 4], IN[t.base + 5],
           nullptr, nullptr, nullptr, nullptr, hB,
           B, t.cin, 4096, t.cout, 4096, 3, 1, t.d, t.d, 1);
    float*  oF = alloc((size_t)B * t.cout * 4096);
    __bf16* oB = allocB((size_t)B * t.cout * 4096);
    conv1d(hB, IN[t.base + 6], IN[t.base + 7],
           IN[t.base + 8], IN[t.base + 9], IN[t.base + 10], IN[t.base + 11],
           nullptr, nullptr, res, oF, oB,
           B, t.cout, 4096, t.cout, 4096, 3, 1, t.d, t.d, 2);
    curF = oF; curB = oB;
  }
  mean_kernel<<<gridFor(B * 512), 256, 0, stream>>>(curF, zpre, B * 512, 4096);
  linear_kernel<<<gridFor(B * 512), 256, 0, stream>>>(zpre, IN[65], IN[66], z,
                                                      B, 512, 512, 0, 0.f, 0.f);

  // ================= parametric decoder =================
  linear_kernel<<<gridFor(B * 256), 256, 0, stream>>>(z,   IN[67], IN[68], ph1, B, 512, 256, 1, 0.f, 0.f);
  linear_kernel<<<gridFor(B * 256), 256, 0, stream>>>(ph1, IN[69], IN[70], ph2, B, 256, 256, 1, 0.f, 0.f);
  linear_kernel<<<1, 64, 0, stream>>>(ph2, IN[71], IN[72], freqs, B, 256, 5, 2, 19980.f, 20.f);
  linear_kernel<<<1, 64, 0, stream>>>(ph2, IN[73], IN[74], gains, B, 256, 5, 3, 12.f, 0.f);
  linear_kernel<<<1, 64, 0, stream>>>(ph2, IN[75], IN[76], qsb,   B, 256, 5, 2, 4.5f, 0.5f);
  eq_coef_kernel<<<1, 64, 0, stream>>>(freqs, gains, qsb, coefs, B * 5);
  biquad_cascade_kernel<<<1, 32, 0, stream>>>(audio, coefs, eq, B, T);

  for (int i = 0; i < 7; ++i) {
    linear_kernel<<<gridFor(B * wubC[i]), 256, 0, stream>>>(
        z, IN[wubBase[i] + 6], IN[wubBase[i] + 7], filmG[i], B, 512, wubC[i], 0, 0.f, 0.f);
    linear_kernel<<<gridFor(B * wubC[i]), 256, 0, stream>>>(
        z, IN[wubBase[i] + 8], IN[wubBase[i] + 9], filmB[i], B, 512, wubC[i], 0, 0.f, 0.f);
  }

  // ================= residual decoder (reuse encoder arena) =================
  off = base0;
  auto wub = [&](const __bf16* x, float* yf, int blk, int cin, int cout, int t) {
    const int bse = wubBase[blk];
    conv1d(x, IN[bse], IN[bse + 1], IN[bse + 2], IN[bse + 3], IN[bse + 4], IN[bse + 5],
           filmG[blk], filmB[blk], nullptr, yf, nullptr, B, cin, t, cout, t, 15, 1, 1, 7, 1);
  };

  float* d1 = alloc((size_t)B * 32 * 65536);  wub(audioB, d1, 0, 2, 32, 65536);
  __bf16* p1 = allocB((size_t)B * 32 * 32768);
  avgpool2_kernel<<<gridFor((long)B * 32 * 32768), 256, 0, stream>>>(d1, p1, (long)B * 32 * 32768, 32768);
  float* d2 = alloc((size_t)B * 64 * 32768);  wub(p1, d2, 1, 32, 64, 32768);
  __bf16* p2 = allocB((size_t)B * 64 * 16384);
  avgpool2_kernel<<<gridFor((long)B * 64 * 16384), 256, 0, stream>>>(d2, p2, (long)B * 64 * 16384, 16384);
  float* d3 = alloc((size_t)B * 128 * 16384); wub(p2, d3, 2, 64, 128, 16384);
  __bf16* p3 = allocB((size_t)B * 128 * 8192);
  avgpool2_kernel<<<gridFor((long)B * 128 * 8192), 256, 0, stream>>>(d3, p3, (long)B * 128 * 8192, 8192);
  float* bt = alloc((size_t)B * 256 * 8192);  wub(p3, bt, 3, 128, 256, 8192);

  __bf16* cat3 = allocB((size_t)B * 384 * 16384);
  interp_concat_kernel<<<gridFor((long)B * 256 * 16384), 256, 0, stream>>>(
      bt, cat3, (long)B * 256 * 16384, 256, 8192, 16384, 384, 0);
  interp_concat_kernel<<<gridFor((long)B * 128 * 16384), 256, 0, stream>>>(
      d3, cat3, (long)B * 128 * 16384, 128, 16384, 16384, 384, 256);
  float* u3 = alloc((size_t)B * 128 * 16384); wub(cat3, u3, 4, 384, 128, 16384);

  __bf16* cat2 = allocB((size_t)B * 192 * 32768);
  interp_concat_kernel<<<gridFor((long)B * 128 * 32768), 256, 0, stream>>>(
      u3, cat2, (long)B * 128 * 32768, 128, 16384, 32768, 192, 0);
  interp_concat_kernel<<<gridFor((long)B * 64 * 32768), 256, 0, stream>>>(
      d2, cat2, (long)B * 64 * 32768, 64, 32768, 32768, 192, 128);
  float* u2 = alloc((size_t)B * 64 * 32768);  wub(cat2, u2, 5, 192, 64, 32768);

  __bf16* cat1 = allocB((size_t)B * 96 * 65536);
  interp_concat_kernel<<<gridFor((long)B * 64 * 65536), 256, 0, stream>>>(
      u2, cat1, (long)B * 64 * 65536, 64, 32768, 65536, 96, 0);
  interp_concat_kernel<<<gridFor((long)B * 32 * 65536), 256, 0, stream>>>(
      d1, cat1, (long)B * 32 * 65536, 32, 65536, 65536, 96, 64);
  float* u1 = alloc((size_t)B * 32 * 65536);  wub(cat1, u1, 6, 96, 32, 65536);

  float* ro = alloc((size_t)B * 2 * 65536);
  pconv_kernel<<<gridFor((long)B * 2 * 65536), 256, 0, stream>>>(
      u1, IN[147], IN[148], ro, (long)B * 2 * 65536, 32, 2, 65536);

  add_kernel<<<gridFor((long)B * 2 * T), 256, 0, stream>>>(eq, ro, outp, (long)B * 2 * T);
}
